// MaskedAttention_19533511262815
// MI455X (gfx1250) — matrix operand order repaired and checked by reading, not compiled, not run on hardware
//
#include <hip/hip_runtime.h>
#include <stdint.h>

// Problem sizes (fixed by the reference)
#define B_    2
#define SEQ   4096
#define DIM   1024
#define NHEAD 16
#define DH    64
#define N3    3072  // 3*DIM
// softmax scale 1/sqrt(64) folded together with log2(e) into Q at projection
// time, so the online softmax uses exp2 (single v_exp_f32).
#define QK_SCALE (0.125f * 1.4426950408889634f)

typedef __bf16 bf16_t;
typedef __attribute__((ext_vector_type(16))) __bf16 v16bf;
typedef __attribute__((ext_vector_type(8)))  __bf16 bf16x8;
typedef __attribute__((ext_vector_type(8)))  float  v8f;

union ABPack { v16bf v; bf16x8 h[2]; };

static __device__ __forceinline__ bf16_t f2bf(float f) {
  unsigned int u = __builtin_bit_cast(unsigned int, f);
  unsigned int r = (u + 0x7FFFu + ((u >> 16) & 1u)) >> 16;  // RNE
  return __builtin_bit_cast(bf16_t, (unsigned short)r);
}

static __device__ __forceinline__ v8f wmma_bf16(v16bf a, v16bf b, v8f c) {
  return __builtin_amdgcn_wmma_f32_16x16x32_bf16(false, a, false, b, (short)0, c,
                                                 false, false);
}

// A operand: 16(M) x 32(K) bf16 from row-major [M x lda].
static __device__ __forceinline__ v16bf load_A(const bf16_t* base, int lda, int lane) {
  int row = lane & 15, grp = lane >> 4;
  ABPack u;
  u.h[0] = *(const bf16x8*)(base + (size_t)row * lda + grp * 8);
  u.h[1] = *(const bf16x8*)(base + (size_t)row * lda + 16 + grp * 8);
  return u.v;
}

// B operand: 32(K) x 16(N) bf16 from an N-major source ("B^T").
static __device__ __forceinline__ v16bf load_BT(const bf16_t* base, int ldb, int lane) {
  int col = lane & 15, grp = lane >> 4;
  ABPack u;
  u.h[0] = *(const bf16x8*)(base + (size_t)col * ldb + grp * 8);
  u.h[1] = *(const bf16x8*)(base + (size_t)col * ldb + 16 + grp * 8);
  return u.v;
}

static __device__ __forceinline__ float redmax16(float v) {
  v = fmaxf(v, __shfl_xor(v, 1, 32));
  v = fmaxf(v, __shfl_xor(v, 2, 32));
  v = fmaxf(v, __shfl_xor(v, 4, 32));
  v = fmaxf(v, __shfl_xor(v, 8, 32));
  return v;
}
static __device__ __forceinline__ float redsum16(float v) {
  v += __shfl_xor(v, 1, 32);
  v += __shfl_xor(v, 2, 32);
  v += __shfl_xor(v, 4, 32);
  v += __shfl_xor(v, 8, 32);
  return v;
}

// ---- CDNA5 async memory->LDS copy (ASYNCcnt-tracked), inline asm ----
// LDS byte address = low 32 bits of the generic pointer (HW truncates flat
// addresses into LDS space).
static __device__ __forceinline__ void async_copy_b128(void* lds_ptr, const void* gptr) {
  unsigned int lds_off = (unsigned int)(uintptr_t)lds_ptr;
  unsigned long long ga = (unsigned long long)(uintptr_t)gptr;
  asm volatile("global_load_async_to_lds_b128 %0, %1, off"
               :: "v"(lds_off), "v"(ga) : "memory");
}
static __device__ __forceinline__ void wait_async0() {
  asm volatile("s_wait_asynccnt 0x0" ::: "memory");
}

// ---------------- conversion kernels ----------------
__global__ void k_conv_x(const float* __restrict__ x, bf16_t* __restrict__ xbf) {
  size_t i = (size_t)blockIdx.x * 256 + threadIdx.x;  // exact grid
  xbf[i] = f2bf(x[i]);
}
__global__ void k_conv_wqkv(const float* __restrict__ w, bf16_t* __restrict__ wT) {
  size_t i = (size_t)blockIdx.x * 256 + threadIdx.x;  // over DIM*N3
  int k = (int)(i / N3);
  int n = (int)(i % N3);
  wT[(size_t)n * DIM + k] = f2bf(w[i]);
}
__global__ void k_conv_wout(const float* __restrict__ w, bf16_t* __restrict__ wT) {
  size_t i = (size_t)blockIdx.x * 256 + threadIdx.x;  // over DIM*DIM
  int k = (int)(i >> 10);
  int n = (int)(i & 1023);
  wT[(size_t)n * DIM + k] = f2bf(w[i]);
}

// ---------------- QKV projection GEMM ----------------
// [8192 x 1024] @ [1024 x 3072]; per-wave tile 32(M) x 64(N).
// K-loop unrolled 2x with ping-pong operand register sets: loads for the next
// half-step land directly in the registers the next WMMA block consumes (no
// rotation movs, no WMMA->VALU WAR hazard nops).
__global__ __launch_bounds__(256) void k_qkv_gemm(
    const bf16_t* __restrict__ xbf, const bf16_t* __restrict__ wT,
    bf16_t* __restrict__ Qbf, bf16_t* __restrict__ Kbf, bf16_t* __restrict__ Vt) {
  int lane = threadIdx.x & 31, wid = threadIdx.x >> 5;
  int gw = blockIdx.x * 8 + wid;            // 12288 waves total
  int nt = gw % (N3 / 64);                  // 48 N tiles
  int mt = gw / (N3 / 64);                  // 256 M tiles
  int m0 = mt * 32, n0 = nt * 64;

  const bf16_t* arow0 = xbf + (size_t)m0 * DIM;
  const bf16_t* arow1 = xbf + (size_t)(m0 + 16) * DIM;
  const bf16_t* wTn = wT + (size_t)n0 * DIM;

  v8f acc0[4] = {}, acc1[4] = {};
  v16bf xa0 = load_A(arow0, DIM, lane);
  v16bf xa1 = load_A(arow1, DIM, lane);
  v16bf xb[4];
#pragma unroll
  for (int s = 0; s < 4; ++s) xb[s] = load_BT(wTn + (size_t)s * 16 * DIM, DIM, lane);

  int k = 0;
  for (; k < DIM - 64; k += 64) {
    v16bf ya0 = load_A(arow0 + k + 32, DIM, lane);
    v16bf ya1 = load_A(arow1 + k + 32, DIM, lane);
    v16bf yb[4];
#pragma unroll
    for (int s = 0; s < 4; ++s)
      yb[s] = load_BT(wTn + (size_t)s * 16 * DIM + k + 32, DIM, lane);
#pragma unroll
    for (int s = 0; s < 4; ++s) acc0[s] = wmma_bf16(xa0, xb[s], acc0[s]);
#pragma unroll
    for (int s = 0; s < 4; ++s) acc1[s] = wmma_bf16(xa1, xb[s], acc1[s]);
    xa0 = load_A(arow0 + k + 64, DIM, lane);
    xa1 = load_A(arow1 + k + 64, DIM, lane);
#pragma unroll
    for (int s = 0; s < 4; ++s)
      xb[s] = load_BT(wTn + (size_t)s * 16 * DIM + k + 64, DIM, lane);
#pragma unroll
    for (int s = 0; s < 4; ++s) acc0[s] = wmma_bf16(ya0, yb[s], acc0[s]);
#pragma unroll
    for (int s = 0; s < 4; ++s) acc1[s] = wmma_bf16(ya1, yb[s], acc1[s]);
  }
  {  // tail: k == DIM-64
    v16bf ya0 = load_A(arow0 + k + 32, DIM, lane);
    v16bf ya1 = load_A(arow1 + k + 32, DIM, lane);
    v16bf yb[4];
#pragma unroll
    for (int s = 0; s < 4; ++s)
      yb[s] = load_BT(wTn + (size_t)s * 16 * DIM + k + 32, DIM, lane);
#pragma unroll
    for (int s = 0; s < 4; ++s) acc0[s] = wmma_bf16(xa0, xb[s], acc0[s]);
#pragma unroll
    for (int s = 0; s < 4; ++s) acc1[s] = wmma_bf16(xa1, xb[s], acc1[s]);
#pragma unroll
    for (int s = 0; s < 4; ++s) acc0[s] = wmma_bf16(ya0, yb[s], acc0[s]);
#pragma unroll
    for (int s = 0; s < 4; ++s) acc1[s] = wmma_bf16(ya1, yb[s], acc1[s]);
  }

  // Epilogue: scatter to Q (scaled), K, or V-transposed.
  int sec  = n0 >> 10;            // 0:Q 1:K 2:V
  int head = (n0 & 1023) >> 6;
  int rowOff = (lane < 16) ? 0 : 8;
  int colLane = lane & 15;
#pragma unroll
  for (int part = 0; part < 2; ++part) {
#pragma unroll
    for (int s = 0; s < 4; ++s) {
      int d = s * 16 + colLane;
#pragma unroll
      for (int r = 0; r < 8; ++r) {
        int gi = m0 + part * 16 + rowOff + r;
        int bb = gi >> 12;
        int ii = gi & (SEQ - 1);
        int bh = bb * NHEAD + head;
        float val = part ? acc1[s][r] : acc0[s][r];
        if (sec == 0)
          Qbf[((size_t)bh * SEQ + ii) * DH + d] = f2bf(val * QK_SCALE);
        else if (sec == 1)
          Kbf[((size_t)bh * SEQ + ii) * DH + d] = f2bf(val);
        else
          Vt[((size_t)bh * DH + d) * SEQ + ii] = f2bf(val);
      }
    }
  }
}

// ---------------- flash attention ----------------
// All 8 waves of a block share one (b,h) and walk the same 32-key blocks, so
// K/V tiles are staged in LDS with async global->LDS copies (double-buffered
// one iteration ahead, ASYNCcnt + s_barrier for ordering). Each wave owns 16
// query rows; B operands are read from LDS (ds_load_b128).
__global__ __launch_bounds__(256) void k_attn(
    const bf16_t* __restrict__ Qbf, const bf16_t* __restrict__ Kbf,
    const bf16_t* __restrict__ Vt, const unsigned char* __restrict__ mask,
    bf16_t* __restrict__ Obf) {
  __shared__ __align__(16) bf16_t Kl[2][32][64];    // [buf][key][dh]   4KB each
  __shared__ __align__(16) bf16_t Vl[2][64][32];    // [buf][dh][key]   4KB each
  __shared__ __align__(16) bf16_t pbuf[8][16][32];  // per-wave P bounce

  int tid = threadIdx.x;
  int lane = tid & 31, wid = tid >> 5;
  int gw = blockIdx.x * 8 + wid;          // 8192 waves
  int it = gw & (SEQ / 16 - 1);
  int bh = gw >> 8;                       // uniform within a block
  int b = bh >> 4, h = bh & (NHEAD - 1);
  int i0 = it * 16;

  const bf16_t* Qh = Qbf + (size_t)bh * SEQ * DH;
  const bf16_t* Kh = Kbf + (size_t)bh * SEQ * DH;
  const bf16_t* Vh = Vt + (size_t)bh * DH * SEQ;
  const unsigned char* mrow = mask + (size_t)b * SEQ * SEQ;

  // per-thread async-copy slices: one B128 per thread per tile (256*16B = 4KB)
  int krow = tid >> 3, kseg = tid & 7;    // K tile: 32 rows x 128B
  int vrow = tid >> 2, vseg = tid & 3;    // V tile: 64 rows x 64B

  v16bf qa0 = load_A(Qh + (size_t)i0 * DH, DH, lane);
  v16bf qa1 = load_A(Qh + (size_t)i0 * DH + 32, DH, lane);

  v8f o[4] = {};
  float mrun[8], lrun[8];
#pragma unroll
  for (int r = 0; r < 8; ++r) { mrun[r] = -1e30f; lrun[r] = 0.0f; }

  int rowOff = (lane < 16) ? 0 : 8;
  int colLane = lane & 15;
  int grp = lane >> 4;

  // prologue: stage first key block into buffer 0
  async_copy_b128(&Kl[0][krow][kseg * 8], Kh + (size_t)krow * DH + kseg * 8);
  async_copy_b128(&Vl[0][vrow][vseg * 8], Vh + (size_t)vrow * SEQ + vseg * 8);
  wait_async0();
  __syncthreads();

  for (int j0 = 0; j0 < SEQ; j0 += 32) {
    int cur = (j0 >> 5) & 1, nxt = cur ^ 1;
    // stage next key block while computing on the current one
    if (j0 + 32 < SEQ) {
      async_copy_b128(&Kl[nxt][krow][kseg * 8],
                      Kh + (size_t)(j0 + 32 + krow) * DH + kseg * 8);
      async_copy_b128(&Vl[nxt][vrow][vseg * 8],
                      Vh + (size_t)vrow * SEQ + (j0 + 32) + vseg * 8);
    }
    // feed L2 two key blocks ahead (global_prefetch_b8)
    if (j0 + 64 < SEQ) {
      __builtin_prefetch(Kh + (size_t)(j0 + 64 + lane) * DH, 0, 3);
      __builtin_prefetch(Vh + (size_t)(2 * lane) * SEQ + j0 + 64, 0, 3);
      __builtin_prefetch(Vh + (size_t)(2 * lane + 1) * SEQ + j0 + 64, 0, 3);
    }
    // mask bytes for this 16x32 tile (overlap the WMMAs below)
    unsigned char mbyte0[8], mbyte1[8];
#pragma unroll
    for (int r = 0; r < 8; ++r) {
      int gi = i0 + rowOff + r;
      const unsigned char* mp = mrow + (size_t)gi * SEQ + j0 + colLane;
      mbyte0[r] = mp[0];
      mbyte1[r] = mp[16];
    }

    // ---- S = Q * K^T : B operands straight out of LDS ----
    v8f s0 = {}, s1 = {};
    {
      ABPack u;
#pragma unroll
      for (int t16 = 0; t16 < 2; ++t16) {
#pragma unroll
        for (int c = 0; c < 2; ++c) {
          u.h[0] = *(const bf16x8*)&Kl[cur][t16 * 16 + colLane][c * 32 + grp * 8];
          u.h[1] = *(const bf16x8*)&Kl[cur][t16 * 16 + colLane][c * 32 + 16 + grp * 8];
          v8f& sd = t16 ? s1 : s0;
          sd = wmma_bf16(c ? qa1 : qa0, u.v, sd);
        }
      }
    }

    // ---- mask (mask==1 -> masked out) ----
#pragma unroll
    for (int r = 0; r < 8; ++r) {
      if (mbyte0[r]) s0[r] = -3.0e38f;
      if (mbyte1[r]) s1[r] = -3.0e38f;
    }

    // ---- online softmax (log2 domain) ----
    float corr[8];
#pragma unroll
    for (int r = 0; r < 8; ++r) {
      float rmax = redmax16(fmaxf(s0[r], s1[r]));
      float mnew = fmaxf(mrun[r], rmax);
      corr[r] = exp2f(mrun[r] - mnew);
      mrun[r] = mnew;
      float p0 = exp2f(s0[r] - mnew);
      float p1 = exp2f(s1[r] - mnew);
      pbuf[wid][rowOff + r][colLane]      = f2bf(p0);
      pbuf[wid][rowOff + r][16 + colLane] = f2bf(p1);
      lrun[r] = lrun[r] * corr[r] + redsum16(p0 + p1);
    }
#pragma unroll
    for (int s = 0; s < 4; ++s)
#pragma unroll
      for (int r = 0; r < 8; ++r) o[s][r] *= corr[r];

    asm volatile("" ::: "memory");  // keep P writes before P reads

    // ---- P (16x32) as A operand from LDS ----
    ABPack pu;
    pu.h[0] = *(const bf16x8*)&pbuf[wid][colLane][grp * 8];
    pu.h[1] = *(const bf16x8*)&pbuf[wid][colLane][16 + grp * 8];

    // ---- O += P * V : B operands from LDS V tile ----
#pragma unroll
    for (int s = 0; s < 4; ++s) {
      ABPack u;
      u.h[0] = *(const bf16x8*)&Vl[cur][s * 16 + colLane][grp * 8];
      u.h[1] = *(const bf16x8*)&Vl[cur][s * 16 + colLane][16 + grp * 8];
      o[s] = wmma_bf16(pu.v, u.v, o[s]);
    }

    // next buffer must be fully staged, and everyone done reading cur,
    // before the following iteration overwrites buffers.
    wait_async0();
    __syncthreads();
  }

  // ---- finalize: divide by row sums, store [b][n][h*dh] bf16 ----
#pragma unroll
  for (int s = 0; s < 4; ++s)
#pragma unroll
    for (int r = 0; r < 8; ++r) {
      int gi = i0 + rowOff + r;
      float val = o[s][r] / lrun[r];
      Obf[((size_t)b * SEQ + gi) * DIM + h * DH + s * 16 + colLane] = f2bf(val);
    }
}

// ---------------- output projection GEMM + bias ----------------
__global__ __launch_bounds__(256) void k_out_gemm(
    const bf16_t* __restrict__ Obf, const bf16_t* __restrict__ wT,
    const float* __restrict__ b_out, float* __restrict__ out) {
  int lane = threadIdx.x & 31, wid = threadIdx.x >> 5;
  int gw = blockIdx.x * 8 + wid;          // 4096 waves
  int nt = gw & 15;
  int mt = gw >> 4;
  int m0 = mt * 32, n0 = nt * 64;

  const bf16_t* arow0 = Obf + (size_t)m0 * DIM;
  const bf16_t* arow1 = Obf + (size_t)(m0 + 16) * DIM;
  const bf16_t* wTn = wT + (size_t)n0 * DIM;

  v8f acc0[4] = {}, acc1[4] = {};
  v16bf xa0 = load_A(arow0, DIM, lane);
  v16bf xa1 = load_A(arow1, DIM, lane);
  v16bf xb[4];
#pragma unroll
  for (int s = 0; s < 4; ++s) xb[s] = load_BT(wTn + (size_t)s * 16 * DIM, DIM, lane);

  int k = 0;
  for (; k < DIM - 64; k += 64) {
    v16bf ya0 = load_A(arow0 + k + 32, DIM, lane);
    v16bf ya1 = load_A(arow1 + k + 32, DIM, lane);
    v16bf yb[4];
#pragma unroll
    for (int s = 0; s < 4; ++s)
      yb[s] = load_BT(wTn + (size_t)s * 16 * DIM + k + 32, DIM, lane);
#pragma unroll
    for (int s = 0; s < 4; ++s) acc0[s] = wmma_bf16(xa0, xb[s], acc0[s]);
#pragma unroll
    for (int s = 0; s < 4; ++s) acc1[s] = wmma_bf16(xa1, xb[s], acc1[s]);
    xa0 = load_A(arow0 + k + 64, DIM, lane);
    xa1 = load_A(arow1 + k + 64, DIM, lane);
#pragma unroll
    for (int s = 0; s < 4; ++s)
      xb[s] = load_BT(wTn + (size_t)s * 16 * DIM + k + 64, DIM, lane);
#pragma unroll
    for (int s = 0; s < 4; ++s) acc0[s] = wmma_bf16(ya0, yb[s], acc0[s]);
#pragma unroll
    for (int s = 0; s < 4; ++s) acc1[s] = wmma_bf16(ya1, yb[s], acc1[s]);
  }
  {
    v16bf ya0 = load_A(arow0 + k + 32, DIM, lane);
    v16bf ya1 = load_A(arow1 + k + 32, DIM, lane);
    v16bf yb[4];
#pragma unroll
    for (int s = 0; s < 4; ++s)
      yb[s] = load_BT(wTn + (size_t)s * 16 * DIM + k + 32, DIM, lane);
#pragma unroll
    for (int s = 0; s < 4; ++s) acc0[s] = wmma_bf16(xa0, xb[s], acc0[s]);
#pragma unroll
    for (int s = 0; s < 4; ++s) acc1[s] = wmma_bf16(xa1, xb[s], acc1[s]);
#pragma unroll
    for (int s = 0; s < 4; ++s) acc0[s] = wmma_bf16(ya0, yb[s], acc0[s]);
#pragma unroll
    for (int s = 0; s < 4; ++s) acc1[s] = wmma_bf16(ya1, yb[s], acc1[s]);
  }

  int rowOff = (lane < 16) ? 0 : 8;
  int colLane = lane & 15;
#pragma unroll
  for (int s = 0; s < 4; ++s) {
    int n = n0 + s * 16 + colLane;
    float bias = b_out[n];
#pragma unroll
    for (int r = 0; r < 8; ++r) {
      int gi0 = m0 + rowOff + r;
      out[(size_t)gi0 * DIM + n] = acc0[s][r] + bias;
      int gi1 = m0 + 16 + rowOff + r;
      out[(size_t)gi1 * DIM + n] = acc1[s][r] + bias;
    }
  }
}

extern "C" void kernel_launch(void* const* d_in, const int* in_sizes, int n_in,
                              void* d_out, int out_size, void* d_ws, size_t ws_size,
                              hipStream_t stream) {
  (void)in_sizes; (void)n_in; (void)out_size; (void)ws_size;
  const float* x = (const float*)d_in[0];
  const unsigned char* mask = (const unsigned char*)d_in[1];  // numpy bool, 1B
  const float* w_qkv = (const float*)d_in[2];
  const float* w_out = (const float*)d_in[3];
  const float* b_out = (const float*)d_in[4];
  float* out = (float*)d_out;

  char* ws = (char*)d_ws;
  size_t off = 0;
  bf16_t* xbf   = (bf16_t*)(ws + off); off += (size_t)B_ * SEQ * DIM * 2;
  bf16_t* wqkvT = (bf16_t*)(ws + off); off += (size_t)N3 * DIM * 2;
  bf16_t* woutT = (bf16_t*)(ws + off); off += (size_t)DIM * DIM * 2;
  bf16_t* Qbf   = (bf16_t*)(ws + off); off += (size_t)B_ * NHEAD * SEQ * DH * 2;
  bf16_t* Kbf   = (bf16_t*)(ws + off); off += (size_t)B_ * NHEAD * SEQ * DH * 2;
  bf16_t* Vt    = (bf16_t*)(ws + off); off += (size_t)B_ * NHEAD * DH * SEQ * 2;
  bf16_t* Obf   = (bf16_t*)(ws + off); off += (size_t)B_ * SEQ * DIM * 2;

  k_conv_x<<<(B_ * SEQ * DIM) / 256, 256, 0, stream>>>(x, xbf);
  k_conv_wqkv<<<(DIM * N3) / 256, 256, 0, stream>>>(w_qkv, wqkvT);
  k_conv_wout<<<(DIM * DIM) / 256, 256, 0, stream>>>(w_out, woutT);

  k_qkv_gemm<<<1536, 256, 0, stream>>>(xbf, wqkvT, Qbf, Kbf, Vt);
  k_attn<<<1024, 256, 0, stream>>>(Qbf, Kbf, Vt, mask, Obf);
  k_out_gemm<<<512, 256, 0, stream>>>(Obf, woutT, b_out, out);
}
